// WeightedGNNModel_70832600646081
// MI455X (gfx1250) — matrix operand order repaired and checked by reading, not compiled, not run on hardware
//
#include <hip/hip_runtime.h>

typedef __attribute__((ext_vector_type(16))) _Float16 v16h;
typedef __attribute__((ext_vector_type(8)))  float    v8f;

#define ALPHA_F 0.3f
#define NSLICES 64

// ---------------------------------------------------------------- fill
__global__ void fill_f32(float* __restrict__ p, float v, long long n) {
    long long i = (long long)blockIdx.x * blockDim.x + threadIdx.x;
    long long stride = (long long)gridDim.x * blockDim.x;
    for (; i < n; i += stride) p[i] = v;
}

// ---------------------------------------------------------------- degree
__global__ void deg_accum(const long long* __restrict__ col,
                          const float* __restrict__ w,
                          float* __restrict__ deg, int nE) {
    int i = blockIdx.x * blockDim.x + threadIdx.x;
    int stride = gridDim.x * blockDim.x;
    for (; i < nE; i += stride) {
        atomicAdd(&deg[col[i]], w[i]);
    }
}

__global__ void deg_to_dinv(float* __restrict__ d, int nN) {
    int i = blockIdx.x * blockDim.x + threadIdx.x;
    if (i < nN) {
        float v = d[i];
        d[i] = (v > 0.0f) ? rsqrtf(v) : 0.0f;
    }
}

// ---------------------------------------------------------------- edge scatter
// Wave handles a tile of 32 edges: each lane loads metadata for ONE edge and
// computes its norm (so row/col/w/dinv are loaded once, not 32x), then the
// inner loop broadcasts one edge at a time via lane shuffles; all 32 lanes
// cooperate on the 64-wide feature row (float2 gather + 2 atomic f32 adds).
// h and agg are L2-resident (25.6 MB each << 192 MB L2).
__global__ void edge_agg(const long long* __restrict__ row,
                         const long long* __restrict__ col,
                         const float* __restrict__ w,
                         const float* __restrict__ dinv,
                         const float* __restrict__ h,
                         float* __restrict__ agg, int nE) {
    int lane   = threadIdx.x & 31;
    int wave   = blockIdx.x * (blockDim.x >> 5) + (threadIdx.x >> 5);
    int nWaves = gridDim.x * (blockDim.x >> 5);

    for (int base = wave * 32; base < nE; base += nWaves * 32) {
        int eL = base + lane;
        int ec = min(eL, nE - 1);              // clamped; tail lanes harmless
        int rl = (int)row[ec];                 // node ids fit in int
        int cl = (int)col[ec];
        float nl = dinv[rl] * w[ec] * dinv[cl];
        int cnt = min(32, nE - base);
        for (int j = 0; j < cnt; ++j) {        // wave-uniform trip count
            int   r   = __shfl(rl, j, 32);
            int   c   = __shfl(cl, j, 32);
            float nrm = __shfl(nl, j, 32);
            float2 v = ((const float2*)(h + (long long)r * 64))[lane];
            float* dst = agg + (long long)c * 64 + lane * 2;
            atomicAdd(dst,     nrm * v.x);
            atomicAdd(dst + 1, nrm * v.y);
        }
    }
}

// ---------------------------------------------------------------- node update
// agg[i] holds sum over non-self edges; add self-loop (dinv^2 * h), APPNP mix,
// sigmoid, write back in place (agg becomes the next layer's h).
__global__ void node_update(const float* __restrict__ hin,
                            float* __restrict__ agg,
                            const float* __restrict__ dinv, int nN) {
    long long total = (long long)nN * 64;
    long long i = (long long)blockIdx.x * blockDim.x + threadIdx.x;
    long long stride = (long long)gridDim.x * blockDim.x;
    for (; i < total; i += stride) {
        int node = (int)(i >> 6);
        float di = dinv[node];
        float h  = hin[i];
        float a  = agg[i] + di * di * h;
        float y  = (1.0f - ALPHA_F) * a + ALPHA_F * h;
        agg[i] = 1.0f / (1.0f + __expf(-y));
    }
}

// ---------------------------------------------------------------- WMMA pool
// sums[g][f] = sum_n [batch[n]==g] * h[n][f]  as a 64 x nN x 64 GEMM on
// v_wmma_f32_16x16x32_f16. blockDim=128: wave w handles f-tile w; blockIdx
// encodes (g-tile, K-slice). batch is sorted -> wave-uniform chunk skip keeps
// EXEC all-ones around the WMMA (ISA requirement) and prunes empty tiles.
// All loads are branchless (clamped addresses); OOB K rows are masked by the
// zero one-hot column in A, so B may read clamped garbage safely.
__global__ void pool_wmma(const float* __restrict__ h,
                          const long long* __restrict__ batch,
                          float* __restrict__ sums, int nN) {
    int lane  = threadIdx.x & 31;
    int ftile = threadIdx.x >> 5;       // 0..3
    int gtile = blockIdx.x & 3;         // 0..3
    int slice = blockIdx.x >> 2;        // 0..NSLICES-1
    int gbase = gtile * 16;
    int fbase = ftile * 16;
    int m   = lane & 15;                // row within A tile
    int hi  = lane >> 4;                // half-wave selector
    int col = lane & 15;                // column within B/C tile
    int myg = gbase + m;

    int nChunks = (nN + 31) >> 5;
    v8f acc = {};
    bool any = false;

    for (int chunk = slice; chunk < nChunks; chunk += NSLICES) {
        int k0   = chunk << 5;
        int kend = min(k0 + 31, nN - 1);
        long long bmin = batch[k0];
        long long bmax = batch[kend];
        if (bmax < (long long)gbase || bmin >= (long long)(gbase + 16))
            continue;                   // wave-uniform: EXEC stays full
        any = true;

        // One clamped batch load per lane; OOB lanes -> -1 via cndmask.
        int kl = min(k0 + lane, nN - 1);
        int bl = (int)batch[kl];
        bl = (k0 + lane < nN) ? bl : -1;

        // A: 16x32 f16 one-hot indicator (contiguous K = 16*hi + i), built from
        // wave shuffles of the single per-lane batch value.
        v16h a;
#pragma unroll
        for (int i = 0; i < 16; ++i) {
            int k  = i + (hi << 4);             // half i -> K index
            int bk = __shfl(bl, k, 32);
            a[i] = (bk == myg) ? (_Float16)1.0f : (_Float16)0.0f;
        }

        // B: 32x16 f16 slice of H (f32->f16), row-striped layout; clamped
        // unconditional loads (masked by A for the tail chunk).
        v16h b;
        int kb = hi << 4;
#pragma unroll
        for (int i = 0; i < 16; ++i) {
            int kk = min(k0 + kb + i, nN - 1);
            b[i] = (_Float16)h[(long long)kk * 64 + fbase + col];
        }

        acc = __builtin_amdgcn_wmma_f32_16x16x32_f16(
                  false, a, false, b, (short)0, acc, false, false);
    }

    if (any) {
        // C layout: VGPR r -> M = r (lanes 0-15) / r+8 (lanes 16-31), N = col.
#pragma unroll
        for (int r = 0; r < 8; ++r) {
            int g = gbase + r + (hi << 3);
            atomicAdd(&sums[g * 64 + fbase + col], acc[r]);
        }
    }
}

// ---------------------------------------------------------------- counts
// LDS histogram (ds_add_f32) then one global atomic per bin per block.
__global__ void count_pool(const long long* __restrict__ batch,
                           float* __restrict__ counts, int nN) {
    __shared__ float bins[64];
    if (threadIdx.x < 64) bins[threadIdx.x] = 0.0f;
    __syncthreads();
    int i = blockIdx.x * blockDim.x + threadIdx.x;
    int stride = gridDim.x * blockDim.x;
    for (; i < nN; i += stride)
        atomicAdd(&bins[(int)batch[i]], 1.0f);
    __syncthreads();
    if (threadIdx.x < 64) {
        float v = bins[threadIdx.x];
        if (v != 0.0f) atomicAdd(&counts[threadIdx.x], v);
    }
}

// ---------------------------------------------------------------- finalize
__global__ void finalize(const float* __restrict__ sums,
                         const float* __restrict__ counts,
                         float* __restrict__ out) {
    int i = blockIdx.x * blockDim.x + threadIdx.x;
    if (i < 64 * 64) {
        float c = counts[i >> 6];
        out[i] = sums[i] / fmaxf(c, 1.0f);
    }
}

// ---------------------------------------------------------------- launch
extern "C" void kernel_launch(void* const* d_in, const int* in_sizes, int n_in,
                              void* d_out, int out_size, void* d_ws, size_t ws_size,
                              hipStream_t stream) {
    (void)n_in; (void)out_size; (void)ws_size;
    const float*     x     = (const float*)d_in[0];
    const long long* ei    = (const long long*)d_in[1];   // int64 (2, E)
    const float*     ew    = (const float*)d_in[2];
    const long long* batch = (const long long*)d_in[3];   // int64, sorted
    int nE = in_sizes[2];
    int nN = in_sizes[3];
    const long long* row = ei;
    const long long* col = ei + nE;

    // Workspace partition (256B aligned): dinv | bufA | bufB | sums | counts
    char*  ws  = (char*)d_ws;
    size_t off = 0;
    auto carve = [&](size_t bytes) {
        void* p = ws + off;
        off = (off + bytes + 255) & ~(size_t)255;
        return p;
    };
    float* dinv   = (float*)carve(sizeof(float) * (size_t)nN);
    float* bufA   = (float*)carve(sizeof(float) * (size_t)nN * 64);
    float* bufB   = (float*)carve(sizeof(float) * (size_t)nN * 64);
    float* sums   = (float*)carve(sizeof(float) * 64 * 64);
    float* counts = (float*)carve(sizeof(float) * 64);

    const int TB = 256;
    long long feat = (long long)nN * 64;

    // ---- degrees: self-loop weight 1, scatter-add edge weights, rsqrt.
    fill_f32<<<(nN + TB - 1) / TB, TB, 0, stream>>>(dinv, 1.0f, nN);
    deg_accum<<<1024, TB, 0, stream>>>(col, ew, dinv, nE);
    deg_to_dinv<<<(nN + TB - 1) / TB, TB, 0, stream>>>(dinv, nN);

    // ---- layer 1: x -> bufA
    fill_f32<<<2048, TB, 0, stream>>>(bufA, 0.0f, feat);
    edge_agg<<<2048, TB, 0, stream>>>(row, col, ew, dinv, x, bufA, nE);
    node_update<<<2048, TB, 0, stream>>>(x, bufA, dinv, nN);

    // ---- layer 2: bufA -> bufB
    fill_f32<<<2048, TB, 0, stream>>>(bufB, 0.0f, feat);
    edge_agg<<<2048, TB, 0, stream>>>(row, col, ew, dinv, bufA, bufB, nE);
    node_update<<<2048, TB, 0, stream>>>(bufA, bufB, dinv, nN);

    // ---- mean pool: WMMA segment-sum + LDS-histogram counts + divide
    fill_f32<<<16, TB, 0, stream>>>(sums, 0.0f, 64 * 64);
    fill_f32<<<1, 64, 0, stream>>>(counts, 0.0f, 64);
    pool_wmma<<<4 * NSLICES, 128, 0, stream>>>(bufB, batch, sums, nN);
    count_pool<<<512, TB, 0, stream>>>(batch, counts, nN);
    finalize<<<16, TB, 0, stream>>>(sums, counts, (float*)d_out);
}
